// RbfInputNet_19842748908184
// MI455X (gfx1250) — matrix operand order repaired and checked by reading, not compiled, not run on hardware
//
#include <hip/hip_runtime.h>
#include <stdint.h>
#include <math.h>

typedef __attribute__((ext_vector_type(16))) __bf16 bf16x16;
typedef __attribute__((ext_vector_type(8)))  float  f32x8;

#define NQ 16384   // fluid particle count (targets)

// ---------------------------------------------------------------------------
// CSR row pointers from sorted target list (np.nonzero order => sorted).
// ---------------------------------------------------------------------------
__global__ void row_ptr_kernel(const int* __restrict__ tgt, int E,
                               int* __restrict__ row_ptr, int N)
{
    int t = blockIdx.x * blockDim.x + threadIdx.x;
    if (t > N) return;
    int lo = 0, hi = E;
    while (lo < hi) {
        int mid = (lo + hi) >> 1;
        if (tgt[mid] < t) lo = mid + 1; else hi = mid;
    }
    row_ptr[t] = lo;
}

// ---------------------------------------------------------------------------
// Per-edge basis: hat(polar(disp)) has <=4 nonzeros of 64. Pack 4 slot ids
// into a u32 and 4 weights into a float4.  sign=-1 for fluid-fluid edges,
// sign=+1 for boundary edges.
// ---------------------------------------------------------------------------
__global__ void edge_basis_kernel(const float* __restrict__ tpos,
                                  const float* __restrict__ spos,
                                  const int*  __restrict__ tgt,
                                  const int*  __restrict__ src,
                                  int E, float sign,
                                  uint32_t* __restrict__ kpack,
                                  float4*   __restrict__ w4)
{
    int e = blockIdx.x * blockDim.x + threadIdx.x;
    if (e >= E) return;
    const float support = sqrtf(16.0f / (3.14159265358979323846f * 16384.0f));
    int t = tgt[e], s = src[e];
    float dx = sign * (spos[2*s+0] - tpos[2*t+0]) / support;
    float dy = sign * (spos[2*s+1] - tpos[2*t+1]) / support;
    dx = fminf(1.f, fmaxf(-1.f, dx));
    dy = fminf(1.f, fmaxf(-1.f, dy));
    float r  = sqrtf(dx*dx + dy*dy + 1e-12f);
    float th = atan2f(dy, dx);
    float m0 = fminf(1.f, fmaxf(-1.f, 2.f*r - 1.f));
    float m1 = fminf(1.f, fmaxf(-1.f, th * (1.0f/3.14159265358979323846f)));
    float px = (m0 + 1.f) * 3.5f;
    float py = (m1 + 1.f) * 3.5f;
    int ix0 = (int)floorf(px); if (ix0 > 7) ix0 = 7; if (ix0 < 0) ix0 = 0;
    int iy0 = (int)floorf(py); if (iy0 > 7) iy0 = 7; if (iy0 < 0) iy0 = 0;
    float fx = px - (float)ix0, fy = py - (float)iy0;
    int ix1 = ix0 + 1 > 7 ? 7 : ix0 + 1;
    int iy1 = iy0 + 1 > 7 ? 7 : iy0 + 1;
    float wx1 = (ix1 != ix0) ? fx : 0.f, wx0 = 1.f - fx;
    float wy1 = (iy1 != iy0) ? fy : 0.f, wy0 = 1.f - fy;
    uint32_t k00 = (uint32_t)(ix0*8 + iy0);
    uint32_t k01 = (uint32_t)(ix0*8 + iy1);
    uint32_t k10 = (uint32_t)(ix1*8 + iy0);
    uint32_t k11 = (uint32_t)(ix1*8 + iy1);
    kpack[e] = k00 | (k01 << 8) | (k10 << 16) | (k11 << 24);
    w4[e] = make_float4(wx0*wy0, wx0*wy1, wx1*wy0, wx1*wy1);
}

// ---------------------------------------------------------------------------
// Weight prep (generic): Wt[coutPad][KtotPad] bf16, K = slot*cin + c, with the
// optional Linear folded in at slot 64.  bias[j] = bc[j] + bl[j].
// ---------------------------------------------------------------------------
__global__ void prep_w_kernel(const float* __restrict__ Wc, const float* __restrict__ bc,
                              const float* __restrict__ Wl, const float* __restrict__ bl,
                              unsigned short* __restrict__ Wt, float* __restrict__ bias,
                              int cin, int cout, int coutPad, int KtotPad)
{
    long idx = (long)blockIdx.x * blockDim.x + threadIdx.x;
    long tot = (long)coutPad * KtotPad;
    if (idx >= tot) return;
    int j = (int)(idx / KtotPad);
    int k = (int)(idx % KtotPad);
    int kconv = 64 * cin;
    float v = 0.f;
    if (j < cout) {
        if (k < kconv) {
            int slot = k / cin, c = k % cin;
            v = Wc[(long)(slot*cin + c)*cout + j];
        } else {
            int c = k - kconv;
            if (Wl && c < cin) v = Wl[(long)c*cout + j];
        }
    }
    ((__bf16*)Wt)[(long)j*KtotPad + k] = (__bf16)v;
    if (k == 0) bias[j] = (j < cout) ? (bc[j] + (Wl ? bl[j] : 0.f)) : 0.f;
}

// Layer-0 fluid launch: cols 0..31 = Linear(Wl0) via slot 64; 32..63 = Conv(Wc0).
__global__ void prep_w0_kernel(const float* __restrict__ Wc0, const float* __restrict__ bc0,
                               const float* __restrict__ Wl0, const float* __restrict__ bl0,
                               unsigned short* __restrict__ Wt, float* __restrict__ bias)
{
    const int Ktot = 544;
    int idx = blockIdx.x * blockDim.x + threadIdx.x;
    if (idx >= 96 * Ktot) return;
    int j = idx / Ktot, k = idx % Ktot;
    float v = 0.f;
    if (k < 512) {
        if (j >= 32 && j < 64) { int slot = k >> 3, c = k & 7; v = Wc0[(slot*8 + c)*32 + (j-32)]; }
    } else if (k < 520) {
        int c = k - 512;
        if (j < 32) v = Wl0[c*32 + j];
    }
    ((__bf16*)Wt)[j*Ktot + k] = (__bf16)v;
    if (k == 0) bias[j] = (j < 32) ? bl0[j] : ((j < 64) ? bc0[j-32] : 0.f);
}

// ---------------------------------------------------------------------------
// Fused continuous-conv block (see round analysis).
// ---------------------------------------------------------------------------
__global__ void __launch_bounds__(256)
rbf_block_kernel(const float* __restrict__ x,
                 const float* __restrict__ xt,
                 const float* __restrict__ resid,
                 const int*   __restrict__ row_ptr,
                 const int*   __restrict__ e_src,
                 const uint32_t* __restrict__ kpack,
                 const float4*   __restrict__ w4,
                 const unsigned short* __restrict__ Wt_u16,
                 const float* __restrict__ bias,
                 float* __restrict__ out,
                 int cin, int cout, int nslots, int chunk_slots,
                 int KtotPad, int out_stride, int out_col0,
                 int relu_in, int accumulate, float scale)
{
    extern __shared__ float Gs[];
    const __bf16* Wt = (const __bf16*)Wt_u16;
    const int tile = blockIdx.x;
    const int t0   = tile * 16;
    const int lane = threadIdx.x & 31;
    const int wave = threadIdx.x >> 5;
    const int ntiles = (cout + 15) >> 4;
    const int ksplit = (2 * ntiles <= 8) ? 2 : 1;
    const bool mma_wave = (wave < ntiles * ksplit);
    const int my_tile  = mma_wave ? (wave % ntiles) : 0;
    const int my_phase = mma_wave ? (wave / ntiles) : 0;

    f32x8 acc = {0.f,0.f,0.f,0.f,0.f,0.f,0.f,0.f};

    const int m   = lane & 15;
    const int kb  = (lane >> 4) * 8;
    const int kbb = (lane >> 4) * 16;
    const int jn  = my_tile * 16 + m;

    const int nmain   = 64 / chunk_slots;
    const int nchunks = nmain + (nslots > 64 ? 1 : 0);
    for (int ch = 0; ch < nchunks; ++ch) {
        const bool tail = (ch >= nmain);
        const int s_lo = tail ? 64 : ch * chunk_slots;
        const int slots_here = tail ? 1 : chunk_slots;
        const int ck = slots_here * cin;
        const int ckpad = (ck + 31) & ~31;

        for (int i = threadIdx.x; i < 16 * ckpad; i += 256) Gs[i] = 0.f;
        __syncthreads();

        for (int rr = 0; rr < 2; ++rr) {
            const int row = wave + rr * 8;
            const int t = t0 + row;
            float* g = Gs + row * ckpad;
            if (!tail) {
                const int eb = row_ptr[t], ee = row_ptr[t+1];
                for (int e = eb; e < ee; ++e) {
                    if (e + 4 < ee) __builtin_prefetch(&e_src[e + 4], 0, 1);
                    const int s = e_src[e];
                    const uint32_t kp = kpack[e];
                    const float4 w = w4[e];
                    float xv[3];
                    #pragma unroll
                    for (int q = 0; q < 3; ++q) {
                        int c = lane + q * 32;
                        float v = (c < cin) ? x[(long)s*cin + c] : 0.f;
                        if (relu_in) v = fmaxf(v, 0.f);
                        xv[q] = v;
                    }
                    #pragma unroll
                    for (int b = 0; b < 4; ++b) {
                        const int slot = (int)((kp >> (8*b)) & 255u);
                        if (slot >= s_lo && slot < s_lo + slots_here) {
                            const float wgt = (b==0) ? w.x : (b==1) ? w.y : (b==2) ? w.z : w.w;
                            float* gb = g + (slot - s_lo) * cin;
                            #pragma unroll
                            for (int q = 0; q < 3; ++q) {
                                int c = lane + q * 32;
                                if (c < cin) gb[c] += wgt * xv[q];
                            }
                        }
                    }
                }
            } else if (xt) {
                for (int c = lane; c < cin; c += 32) {
                    float v = xt[(long)t*cin + c];
                    if (relu_in) v = fmaxf(v, 0.f);
                    g[c] = v;
                }
            }
        }
        __syncthreads();

        if (mma_wave) {
            const __bf16* wrow = Wt + (long)jn * KtotPad + (long)s_lo * cin + kbb;
            const float*  grow = Gs + m * ckpad;
            int kk = my_phase * 32;
            if (kk < ckpad) {
                bf16x16 bcur = *(const bf16x16*)(wrow + kk);
                while (kk < ckpad) {
                    const int kn = kk + 32 * ksplit;
                    bf16x16 bnext = bcur;
                    if (kn < ckpad) bnext = *(const bf16x16*)(wrow + kn);
                    float4 g0 = *(const float4*)(grow + kk + kbb);
                    float4 g1 = *(const float4*)(grow + kk + kbb + 4);
                    float4 g2 = *(const float4*)(grow + kk + kbb + 8);
                    float4 g3 = *(const float4*)(grow + kk + kbb + 12);
                    bf16x16 a;
                    a[0]=(__bf16)g0.x; a[1]=(__bf16)g0.y; a[2]=(__bf16)g0.z; a[3]=(__bf16)g0.w;
                    a[4]=(__bf16)g1.x; a[5]=(__bf16)g1.y; a[6]=(__bf16)g1.z; a[7]=(__bf16)g1.w;
                    a[8]=(__bf16)g2.x; a[9]=(__bf16)g2.y; a[10]=(__bf16)g2.z; a[11]=(__bf16)g2.w;
                    a[12]=(__bf16)g3.x; a[13]=(__bf16)g3.y; a[14]=(__bf16)g3.z; a[15]=(__bf16)g3.w;
                    acc = __builtin_amdgcn_wmma_f32_16x16x32_bf16(
                              false, a, false, bcur, (short)0, acc, false, false);
                    bcur = bnext;
                    kk = kn;
                }
            }
        }
        __syncthreads();
    }

    if (ksplit == 2) {
        if (mma_wave && my_phase == 1) {
            float* rs = Gs + my_tile * 256;
            #pragma unroll
            for (int r = 0; r < 8; ++r) rs[r * 32 + lane] = acc[r];
        }
        __syncthreads();
        if (mma_wave && my_phase == 0) {
            const float* rs = Gs + my_tile * 256;
            #pragma unroll
            for (int r = 0; r < 8; ++r) acc[r] += rs[r * 32 + lane];
        }
    }

    if (mma_wave && my_phase == 0) {
        const int mofs = (lane < 16) ? 0 : 8;
        const int j = my_tile * 16 + (lane & 15);
        if (j < cout) {
            #pragma unroll
            for (int r = 0; r < 8; ++r) {
                const int t = t0 + r + mofs;
                float v = acc[r] + bias[j];
                if (resid) v += resid[(long)t * cout + j];
                v *= scale;
                float* p = out + (long)t * out_stride + out_col0 + j;
                if (accumulate) *p += v; else *p = v;
            }
        }
    }
}

// ---------------------------------------------------------------------------
extern "C" void kernel_launch(void* const* d_in, const int* in_sizes, int n_in,
                              void* d_out, int out_size, void* d_ws, size_t ws_size,
                              hipStream_t stream)
{
    (void)n_in; (void)out_size; (void)ws_size;
    const float* fluid_pos  = (const float*)d_in[0];
    const float* bnd_pos    = (const float*)d_in[1];
    const float* fluid_feat = (const float*)d_in[2];
    const float* bnd_feat   = (const float*)d_in[3];
    const int*   f_tgt = (const int*)d_in[4];
    const int*   f_src = (const int*)d_in[5];
    const int*   b_tgt = (const int*)d_in[6];
    const int*   b_src = (const int*)d_in[7];
    const float *Wc0=(const float*)d_in[8],  *bc0=(const float*)d_in[9];
    const float *Wc1=(const float*)d_in[10], *bc1=(const float*)d_in[11];
    const float *Wc2=(const float*)d_in[12], *bc2=(const float*)d_in[13];
    const float *Wc3=(const float*)d_in[14], *bc3=(const float*)d_in[15];
    const float *Wc4=(const float*)d_in[16], *bc4=(const float*)d_in[17];
    const float *Wl0=(const float*)d_in[18], *bl0=(const float*)d_in[19];
    const float *Wl1=(const float*)d_in[20], *bl1=(const float*)d_in[21];
    const float *Wl2=(const float*)d_in[22], *bl2=(const float*)d_in[23];
    const float *Wl3=(const float*)d_in[24], *bl3=(const float*)d_in[25];

    const int N  = NQ;
    const int Ef = in_sizes[4];
    const int Eb = in_sizes[6];

    char* p = (char*)d_ws;
    auto carve = [&](size_t bytes) -> char* {
        char* r = p; p += (bytes + 255) & ~(size_t)255; return r;
    };
    int*      rp_f = (int*)carve((size_t)(N+1)*4);
    int*      rp_b = (int*)carve((size_t)(N+1)*4);
    uint32_t* kp_f = (uint32_t*)carve((size_t)Ef*4);
    float4*   w4_f = (float4*)  carve((size_t)Ef*16);
    uint32_t* kp_b = (uint32_t*)carve((size_t)Eb*4);
    float4*   w4_b = (float4*)  carve((size_t)Eb*16);
    unsigned short* Wt0 = (unsigned short*)carve((size_t)96*544*2);
    float* bias0 = (float*)carve(96*4);
    unsigned short* Wt1 = (unsigned short*)carve((size_t)32*512*2);
    float* bias1 = (float*)carve(32*4);
    unsigned short* Wt2 = (unsigned short*)carve((size_t)64*6240*2);
    float* bias2 = (float*)carve(64*4);
    unsigned short* Wt3 = (unsigned short*)carve((size_t)64*4160*2);
    float* bias3 = (float*)carve(64*4);
    unsigned short* Wt4 = (unsigned short*)carve((size_t)16*4160*2);
    float* bias4 = (float*)carve(16*4);
    float* ans0 = (float*)carve((size_t)N*96*4);
    float* ans1 = (float*)carve((size_t)N*64*4);
    float* ans2 = (float*)carve((size_t)N*64*4);

    row_ptr_kernel<<<(N+256)/256, 256, 0, stream>>>(f_tgt, Ef, rp_f, N);
    row_ptr_kernel<<<(N+256)/256, 256, 0, stream>>>(b_tgt, Eb, rp_b, N);
    edge_basis_kernel<<<(Ef+255)/256, 256, 0, stream>>>(fluid_pos, fluid_pos,
        f_tgt, f_src, Ef, -1.f, kp_f, w4_f);
    edge_basis_kernel<<<(Eb+255)/256, 256, 0, stream>>>(fluid_pos, bnd_pos,
        b_tgt, b_src, Eb, +1.f, kp_b, w4_b);

    prep_w0_kernel<<<(96*544+255)/256, 256, 0, stream>>>(Wc0, bc0, Wl0, bl0, Wt0, bias0);
    prep_w_kernel<<<(32*512+255)/256, 256, 0, stream>>>(Wc1, bc1, nullptr, nullptr,
        Wt1, bias1, 8, 32, 32, 512);
    prep_w_kernel<<<(64*6240+255)/256, 256, 0, stream>>>(Wc2, bc2, Wl1, bl1,
        Wt2, bias2, 96, 64, 64, 6240);
    prep_w_kernel<<<(64*4160+255)/256, 256, 0, stream>>>(Wc3, bc3, Wl2, bl2,
        Wt3, bias3, 64, 64, 64, 4160);
    prep_w_kernel<<<(16*4160+255)/256, 256, 0, stream>>>(Wc4, bc4, Wl3, bl3,
        Wt4, bias4, 64, 2, 16, 4160);

    dim3 grid(N/16), block(256);

    rbf_block_kernel<<<grid, block, 16*512*4, stream>>>(
        fluid_feat, fluid_feat, nullptr, rp_f, f_src, kp_f, w4_f, Wt0, bias0, ans0,
        8, 96, 65, 64, 544, 96, 0, 0, 0, 1.f);

    rbf_block_kernel<<<grid, block, 16*512*4, stream>>>(
        bnd_feat, nullptr, nullptr, rp_b, b_src, kp_b, w4_b, Wt1, bias1, ans0,
        8, 32, 64, 64, 512, 96, 64, 0, 1, 1.f);

    rbf_block_kernel<<<grid, block, 16*768*4, stream>>>(
        ans0, ans0, nullptr, rp_f, f_src, kp_f, w4_f, Wt2, bias2, ans1,
        96, 64, 65, 8, 6240, 64, 0, 1, 0, 1.f);

    rbf_block_kernel<<<grid, block, 16*512*4, stream>>>(
        ans1, ans1, ans1, rp_f, f_src, kp_f, w4_f, Wt3, bias3, ans2,
        64, 64, 65, 8, 4160, 64, 0, 1, 0, 1.f);

    rbf_block_kernel<<<grid, block, 16*512*4, stream>>>(
        ans2, ans2, nullptr, rp_f, f_src, kp_f, w4_f, Wt4, bias4, (float*)d_out,
        64, 2, 65, 8, 4160, 2, 0, 1, 0, 1.0f/128.0f);
}
